// VSSM_42417097015436
// MI455X (gfx1250) — compile-verified, matrix operand order checked by reading, not run on hardware
//
#include <hip/hip_runtime.h>
#include <cstdint>
#include <cstddef>

// ---------------------------------------------------------------------------
// Model constants (from the reference)
// ---------------------------------------------------------------------------
#define BB    8
#define CINc  3
#define IMGc  128
#define DIMc  192
#define DINc  384
#define NST   16
#define RR    12
#define KDIR  4
#define LC    1024
#define MROWS (BB * LC)   // 8192 rows of (b,l)
#define XDBL_W 64         // padded width of x_dbl rows (44 real + 20 zero)

typedef __attribute__((ext_vector_type(16))) _Float16 v16h;
typedef __attribute__((ext_vector_type(8)))  _Float16 v8h;
typedef __attribute__((ext_vector_type(8)))  float    v8f;
typedef __attribute__((ext_vector_type(4)))  unsigned u32x4;
typedef __attribute__((ext_vector_type(8)))  int      i32x8;
typedef __attribute__((ext_vector_type(4)))  int      i32x4;

#if defined(__has_builtin)
#  if __has_builtin(__builtin_amdgcn_tensor_load_to_lds)
#    define HAVE_TDM 1
#  else
#    define HAVE_TDM 0
#  endif
#else
#  define HAVE_TDM 0
#endif
#if __has_include(<hip/amd_detail/amd_gfx1250_TDM.h>)
#  define TDM_6ARG 1        // therock / clang-23 headers -> 6-arg builtin
#else
#  define TDM_6ARG 0        // ROCm 7.2 / clang-22 -> 5-arg builtin
#endif

// ---------------------------------------------------------------------------
// Block-wide sum (wave32 shfl + LDS across waves)
// ---------------------------------------------------------------------------
__device__ inline float block_reduce_sum(float v, float* sred) {
#pragma unroll
  for (int o = 16; o > 0; o >>= 1) v += __shfl_xor(v, o, 32);
  const int lane = threadIdx.x & 31, w = threadIdx.x >> 5;
  const int nw = (blockDim.x + 31) >> 5;
  if (lane == 0) sred[w] = v;
  __syncthreads();
  float t = (threadIdx.x < (unsigned)nw) ? sred[threadIdx.x] : 0.f;
  if (w == 0) {
#pragma unroll
    for (int o = 16; o > 0; o >>= 1) t += __shfl_xor(t, o, 32);
    if (lane == 0) sred[0] = t;
  }
  __syncthreads();
  const float r = sred[0];
  __syncthreads();
  return r;
}

// ---------------------------------------------------------------------------
// fp32 -> fp16 weight conversion (plain, and 44->64 row-padded for x_proj)
// ---------------------------------------------------------------------------
__global__ void cvt_f16(const float* __restrict__ s, _Float16* __restrict__ d, int n) {
  int i = blockIdx.x * blockDim.x + threadIdx.x;
  if (i < n) d[i] = (_Float16)s[i];
}

__global__ void cvt_pad_xp(const float* __restrict__ s, _Float16* __restrict__ d, int total) {
  int i = blockIdx.x * blockDim.x + threadIdx.x;   // over groups*64*384
  if (i >= total) return;
  const int dcol = i % DINc;
  const int rowg = i / DINc;
  const int c = rowg % XDBL_W;                     // padded row within group
  const int g = rowg / XDBL_W;                     // (layer,k) group
  d[i] = (c < 44) ? (_Float16)s[((size_t)g * 44 + c) * DINc + dcol] : (_Float16)0.f;
}

// ---------------------------------------------------------------------------
// Patch embed (conv 4x4 stride 4) + LN  -> y (B,L,192) fp32
// ---------------------------------------------------------------------------
__global__ void patch_embed(const float* __restrict__ x, const float* __restrict__ pw,
                            const float* __restrict__ pb, const float* __restrict__ g,
                            const float* __restrict__ bta, float* __restrict__ y) {
  __shared__ float patch[48];
  __shared__ float sred[16];
  const int bl = blockIdx.x;
  const int b = bl >> 10, l = bl & 1023;
  const int i = l >> 5, j = l & 31;
  const int t = threadIdx.x;
  if (t < 48) {
    const int ci = t >> 4, rem = t & 15, pi = rem >> 2, pj = rem & 3;
    patch[t] = x[(((size_t)b * CINc + ci) * IMGc + (i * 4 + pi)) * IMGc + (j * 4 + pj)];
  }
  __syncthreads();
  float acc = pb[t];
#pragma unroll
  for (int q = 0; q < 48; ++q) acc += patch[q] * pw[t * 48 + q];
  const float mean = block_reduce_sum(acc, sred) * (1.f / DIMc);
  const float d = acc - mean;
  const float var = block_reduce_sum(d * d, sred) * (1.f / DIMc);
  y[(size_t)bl * DIMc + t] = d * rsqrtf(var + 1e-5f) * g[t] + bta[t];
}

// ---------------------------------------------------------------------------
// Row LayerNorm, blockDim == width; OutT = _Float16 or float
// ---------------------------------------------------------------------------
template <typename OutT>
__global__ void ln_rows(const float* __restrict__ X, const float* __restrict__ g,
                        const float* __restrict__ bta, OutT* __restrict__ Y, int W) {
  __shared__ float sred[16];
  const int row = blockIdx.x, t = threadIdx.x;
  const float xv = X[(size_t)row * W + t];
  const float mean = block_reduce_sum(xv, sred) / (float)W;
  const float d = xv - mean;
  const float var = block_reduce_sum(d * d, sred) / (float)W;
  Y[(size_t)row * W + t] = (OutT)(d * rsqrtf(var + 1e-5f) * g[t] + bta[t]);
}

// ---------------------------------------------------------------------------
// WMMA GEMM: C[M,N] (+)= A[M,K] * Bw[N,K]^T  (A,Bw fp16 row-major, C fp32)
// Wave tile = 16(M) x 64(N); M multiple of 128, N multiple of 64 (padded) --
// zero guards, zero divergence, EXEC all-ones at every v_wmma.
// A frag (16x32 f16): lane r=lane&15 -> row, hi=lane>>4 selects K-octets.
// ---------------------------------------------------------------------------
__device__ inline v16h load_frag(const _Float16* p) {
  const v8h lo = *(const v8h*)p;
  const v8h hi = *(const v8h*)(p + 16);
  v16h f;
#pragma unroll
  for (int q = 0; q < 8; ++q) { f[q] = lo[q]; f[q + 8] = hi[q]; }
  return f;
}

template <int ACCUM, int KK>
__global__ void gemm_wmma(const _Float16* __restrict__ A, const _Float16* __restrict__ Bw,
                          float* __restrict__ C, int Nn) {
  const int lane = threadIdx.x & 31;
  const int r = lane & 15, hi = lane >> 4;
  const int wave = threadIdx.x >> 5;
  const int m0 = (blockIdx.x * (blockDim.x >> 5) + wave) * 16;
  const int n0 = blockIdx.y * 64;

  const _Float16* arow = A + (size_t)(m0 + r) * KK;
  const _Float16* b0p = Bw + (size_t)(n0 + r) * KK;
  const _Float16* b1p = b0p + (size_t)16 * KK;
  const _Float16* b2p = b0p + (size_t)32 * KK;
  const _Float16* b3p = b0p + (size_t)48 * KK;

  v8f acc0, acc1, acc2, acc3;
#pragma unroll
  for (int q = 0; q < 8; ++q) { acc0[q] = 0.f; acc1[q] = 0.f; acc2[q] = 0.f; acc3[q] = 0.f; }

#pragma unroll 2
  for (int kk = 0; kk < KK; kk += 32) {
    const int kb = kk + hi * 8;
    const v16h av = load_frag(arow + kb);
    __builtin_prefetch(arow + kb + 32, 0, 1);             // speculative; OOB dropped
    acc0 = __builtin_amdgcn_wmma_f32_16x16x32_f16(false, av, false, load_frag(b0p + kb),
                                                  (short)0, acc0, false, false);
    acc1 = __builtin_amdgcn_wmma_f32_16x16x32_f16(false, av, false, load_frag(b1p + kb),
                                                  (short)0, acc1, false, false);
    acc2 = __builtin_amdgcn_wmma_f32_16x16x32_f16(false, av, false, load_frag(b2p + kb),
                                                  (short)0, acc2, false, false);
    acc3 = __builtin_amdgcn_wmma_f32_16x16x32_f16(false, av, false, load_frag(b3p + kb),
                                                  (short)0, acc3, false, false);
  }

  float* crow = C + (size_t)(m0 + hi * 8) * Nn + n0 + r;  // vgpr j -> row m0+j+hi*8
#pragma unroll
  for (int j = 0; j < 8; ++j) {
    float* cp = crow + (size_t)j * Nn;
    if (ACCUM) {
      cp[0] += acc0[j]; cp[16] += acc1[j]; cp[32] += acc2[j]; cp[48] += acc3[j];
    } else {
      cp[0] = acc0[j]; cp[16] = acc1[j]; cp[32] = acc2[j]; cp[48] = acc3[j];
    }
  }
}

// ---------------------------------------------------------------------------
// Depthwise 3x3 SAME conv + bias + SiLU on xm (first 384 of xz rows).
// Writes xc (B,DIN,L) fp32 and the 4 scan-direction copies xs (K,B,L,DIN) fp16.
// ---------------------------------------------------------------------------
__global__ void conv_silu(const float* __restrict__ xz, const float* __restrict__ cw,
                          const float* __restrict__ cb, float* __restrict__ xc,
                          _Float16* __restrict__ xs) {
  const int bl = blockIdx.x;
  const int b = bl >> 10, l = bl & 1023;
  const int i = l >> 5, j = l & 31;
  const int d = threadIdx.x;
  float acc = cb[d];
#pragma unroll
  for (int di = -1; di <= 1; ++di)
#pragma unroll
    for (int dj = -1; dj <= 1; ++dj) {
      const int ii = i + di, jj = j + dj;
      if (ii >= 0 && ii < 32 && jj >= 0 && jj < 32)
        acc += xz[((size_t)b * LC + ii * 32 + jj) * (2 * DINc) + d] *
               cw[d * 9 + (di + 1) * 3 + (dj + 1)];
    }
  const float v = acc / (1.f + __expf(-acc));             // SiLU
  xc[((size_t)b * DINc + d) * LC + l] = v;
  const int T = ((l & 31) << 5) | (l >> 5);               // spatial transpose of l
  const _Float16 hv = (_Float16)v;
  xs[(((size_t)(0 * BB + b)) * LC + l)          * DINc + d] = hv;   // hw scan
  xs[(((size_t)(1 * BB + b)) * LC + T)          * DINc + d] = hv;   // wh scan
  xs[(((size_t)(2 * BB + b)) * LC + (1023 - l)) * DINc + d] = hv;   // flip hw
  xs[(((size_t)(3 * BB + b)) * LC + (1023 - T)) * DINc + d] = hv;   // flip wh
}

// ---------------------------------------------------------------------------
// Selective scan. One block per (b,k); 384 threads (one per channel d).
// Per step: stage a 44-float (dts[12],B[16],C[16]) row into LDS via the
// Tensor Data Mover (fallback: plain loads), double-buffered, one barrier
// per step. dt = softplus(dts.dtw + dtb); h = exp(dt*A)*h + dt*u*B;
// y = h.C + Ds*u ; scatter to this direction's slot at canonical position.
// ---------------------------------------------------------------------------
__device__ inline void stage_row(const float* __restrict__ gsrc, float* ldst) {
#if HAVE_TDM
  if (threadIdx.x < 32) {                                 // one TDM issue per step
    const unsigned long long ga = (unsigned long long)(uintptr_t)gsrc;
    u32x4 g0;
    g0.x = 1u;                                            // count=1, user mode
    g0.y = (unsigned)(uintptr_t)ldst;                     // lds_addr
    g0.z = (unsigned)ga;                                  // global_addr[31:0]
    g0.w = (unsigned)((ga >> 32) & 0x01FFFFFFu) | (2u << 30);  // addr[56:32] | type=2
    i32x8 g1;
    g1[0] = 2 << 16;                                      // data_size = 4B
    g1[1] = 44 << 16;                                     // tensor_dim0 = 44
    g1[2] = 1 << 16;                                      // tensor_dim1 = 1
    g1[3] = 44 << 16;                                     // tile_dim0 = 44
    g1[4] = 1;                                            // tile_dim1 = 1
    g1[5] = 44; g1[6] = 0; g1[7] = 0;                     // dim0_stride
    i32x4 z4 = {0, 0, 0, 0};
#if TDM_6ARG
    i32x8 z8 = {0, 0, 0, 0, 0, 0, 0, 0};
    __builtin_amdgcn_tensor_load_to_lds(g0, g1, z4, z4, z8, 0);
#else
    __builtin_amdgcn_tensor_load_to_lds(g0, g1, z4, z4, 0);
#endif
  }
#else
  if (threadIdx.x < 44) ldst[threadIdx.x] = gsrc[threadIdx.x];
#endif
}

__device__ inline void stage_wait() {
#if HAVE_TDM
  if (threadIdx.x < 32) __builtin_amdgcn_s_wait_tensorcnt(0);
#endif
}

__global__ void scan_kernel(const float* __restrict__ xdbl, const float* __restrict__ xc,
                            const float* __restrict__ dtw, const float* __restrict__ dtb,
                            const float* __restrict__ alog, const float* __restrict__ dsw,
                            float* __restrict__ yo4) {
  const int b = blockIdx.x >> 2, k = blockIdx.x & 3;
  const int d = threadIdx.x;
  __shared__ float srow[2][48];

  float h[NST], Ar[NST], wr[RR];
#pragma unroll
  for (int n = 0; n < NST; ++n) {
    h[n] = 0.f;
    Ar[n] = -__expf(alog[((size_t)k * DINc + d) * NST + n]);
  }
#pragma unroll
  for (int q = 0; q < RR; ++q) wr[q] = dtw[((size_t)k * DINc + d) * RR + q];
  const float bias = dtb[k * DINc + d];
  const float Dv = dsw[k * DINc + d];

  const float* xrow = xdbl + ((size_t)(k * BB + b)) * LC * XDBL_W;
  const float* xcb = xc + ((size_t)b * DINc + d) * LC;

  stage_row(xrow, &srow[0][0]);
  stage_wait();
  __syncthreads();

  for (int l = 0; l < LC; ++l) {
    const int par = l & 1;
    if (l + 1 < LC) stage_row(xrow + (size_t)(l + 1) * XDBL_W, &srow[par ^ 1][0]);

    float pre = bias;
#pragma unroll
    for (int q = 0; q < RR; ++q) pre += srow[par][q] * wr[q];
    const float dt = (pre > 20.f) ? pre : log1pf(__expf(pre));   // softplus

    const int T = ((l & 31) << 5) | (l >> 5);
    int ml;                                                // canonical position
    if (k == 0) ml = l;
    else if (k == 1) ml = T;
    else if (k == 2) ml = 1023 - l;
    else { const int lf = 1023 - l; ml = ((lf & 31) << 5) | (lf >> 5); }

    const float u = xcb[ml];
    const float du = dt * u;
    float acc = 0.f;
#pragma unroll
    for (int n = 0; n < NST; ++n) {
      h[n] = __expf(dt * Ar[n]) * h[n] + du * srow[par][12 + n];
      acc += h[n] * srow[par][28 + n];
    }
    yo4[(((size_t)b * LC + ml) * DINc + d) * 4 + k] = acc + Dv * u;

    stage_wait();
    __syncthreads();
  }
}

// ---------------------------------------------------------------------------
// Merge 4 directions + out-LN + SiLU(z) gate -> fp16 rows for out_proj
// ---------------------------------------------------------------------------
__global__ void gate_kernel(const float* __restrict__ yo4, const float* __restrict__ xz,
                            const float* __restrict__ g, const float* __restrict__ bta,
                            _Float16* __restrict__ out) {
  __shared__ float sred[16];
  const int bl = blockIdx.x, d = threadIdx.x;
  const float* p = yo4 + ((size_t)bl * DINc + d) * 4;
  const float v = p[0] + p[1] + p[2] + p[3];
  const float mean = block_reduce_sum(v, sred) * (1.f / DINc);
  const float dv = v - mean;
  const float var = block_reduce_sum(dv * dv, sred) * (1.f / DINc);
  const float y = dv * rsqrtf(var + 1e-5f) * g[d] + bta[d];
  const float z = xz[(size_t)bl * (2 * DINc) + DINc + d];
  out[(size_t)bl * DINc + d] = (_Float16)(y * (z / (1.f + __expf(-z))));
}

// ---------------------------------------------------------------------------
// Host launcher
// ---------------------------------------------------------------------------
extern "C" void kernel_launch(void* const* d_in, const int* in_sizes, int n_in,
                              void* d_out, int out_size, void* d_ws, size_t ws_size,
                              hipStream_t stream) {
  (void)in_sizes; (void)n_in; (void)out_size; (void)ws_size;
  const float* x    = (const float*)d_in[0];
  const float* pw   = (const float*)d_in[1];
  const float* pb   = (const float*)d_in[2];
  const float* peg  = (const float*)d_in[3];
  const float* peb  = (const float*)d_in[4];
  const float* lng  = (const float*)d_in[5];
  const float* lnb  = (const float*)d_in[6];
  const float* inw  = (const float*)d_in[7];
  const float* cw   = (const float*)d_in[8];
  const float* cb   = (const float*)d_in[9];
  const float* xpw  = (const float*)d_in[10];
  const float* dtw  = (const float*)d_in[11];
  const float* dtb  = (const float*)d_in[12];
  const float* alog = (const float*)d_in[13];
  const float* dsv  = (const float*)d_in[14];
  const float* ong  = (const float*)d_in[15];
  const float* onb  = (const float*)d_in[16];
  const float* opw  = (const float*)d_in[17];
  const float* fing = (const float*)d_in[18];
  const float* finb = (const float*)d_in[19];
  float* out = (float*)d_out;

  char* ws = (char*)d_ws;
  size_t off = 0;
  auto alloc = [&](size_t bytes) -> void* {
    void* p = ws + off;
    off = (off + bytes + 255) & ~(size_t)255;
    return p;
  };
  float*    f_y    = (float*)alloc((size_t)MROWS * DIMc * 4);
  float*    f_xz   = (float*)alloc((size_t)MROWS * 2 * DINc * 4);
  float*    f_xc   = (float*)alloc((size_t)BB * DINc * LC * 4);
  float*    f_xdbl = (float*)alloc((size_t)KDIR * MROWS * XDBL_W * 4);
  float*    f_yo4  = (float*)alloc((size_t)MROWS * DINc * 4 * 4);
  _Float16* h_yln  = (_Float16*)alloc((size_t)MROWS * DIMc * 2);
  _Float16* h_xs   = (_Float16*)alloc((size_t)KDIR * MROWS * DINc * 2);
  _Float16* h_gate = (_Float16*)alloc((size_t)MROWS * DINc * 2);
  _Float16* h_win  = (_Float16*)alloc((size_t)2 * 768 * 192 * 2);
  _Float16* h_wxp  = (_Float16*)alloc((size_t)2 * KDIR * XDBL_W * DINc * 2);
  _Float16* h_wop  = (_Float16*)alloc((size_t)2 * DIMc * DINc * 2);

  cvt_f16<<<(2 * 768 * 192 + 255) / 256, 256, 0, stream>>>(inw, h_win, 2 * 768 * 192);
  cvt_f16<<<(2 * DIMc * DINc + 255) / 256, 256, 0, stream>>>(opw, h_wop, 2 * DIMc * DINc);
  {
    const int total = 2 * KDIR * XDBL_W * DINc;
    cvt_pad_xp<<<(total + 255) / 256, 256, 0, stream>>>(xpw, h_wxp, total);
  }

  patch_embed<<<MROWS, DIMc, 0, stream>>>(x, pw, pb, peg, peb, f_y);

  // 8 waves per block, each wave owns a 16x64 output tile.
  const int mblocks = MROWS / 16 / 8;   // 64

  for (int layer = 0; layer < 2; ++layer) {
    ln_rows<_Float16><<<MROWS, DIMc, 0, stream>>>(
        f_y, lng + layer * DIMc, lnb + layer * DIMc, h_yln, DIMc);
    // in_proj: 8192 x 768 x 192
    gemm_wmma<0, 192><<<dim3(mblocks, 768 / 64), 256, 0, stream>>>(
        h_yln, h_win + (size_t)layer * 768 * 192, f_xz, 768);
    conv_silu<<<MROWS, DINc, 0, stream>>>(
        f_xz, cw + (size_t)layer * DINc * 9, cb + layer * DINc, f_xc, h_xs);
    // x_proj per direction: 8192 x 64(padded 44) x 384
    for (int k = 0; k < KDIR; ++k)
      gemm_wmma<0, 384><<<dim3(mblocks, 1), 256, 0, stream>>>(
          h_xs + (size_t)k * MROWS * DINc,
          h_wxp + ((size_t)layer * KDIR + k) * XDBL_W * DINc,
          f_xdbl + (size_t)k * MROWS * XDBL_W, XDBL_W);
    scan_kernel<<<BB * KDIR, DINc, 0, stream>>>(
        f_xdbl, f_xc,
        dtw + (size_t)layer * KDIR * DINc * RR, dtb + (size_t)layer * KDIR * DINc,
        alog + (size_t)layer * KDIR * DINc * NST, dsv + (size_t)layer * KDIR * DINc,
        f_yo4);
    gate_kernel<<<MROWS, DINc, 0, stream>>>(
        f_yo4, f_xz, ong + layer * DINc, onb + layer * DINc, h_gate);
    // out_proj (+residual into f_y): 8192 x 192 x 384
    gemm_wmma<1, 384><<<dim3(mblocks, 192 / 64), 256, 0, stream>>>(
        h_gate, h_wop + (size_t)layer * DIMc * DINc, f_y, 192);
  }
  ln_rows<float><<<MROWS, DIMc, 0, stream>>>(f_y, fing, finb, out, DIMc);
}
